// GATLayer_10831907520908
// MI455X (gfx1250) — matrix operand order repaired and checked by reading, not compiled, not run on hardware
//
#include <hip/hip_runtime.h>

typedef __attribute__((ext_vector_type(16))) _Float16 v16h;
typedef __attribute__((ext_vector_type(8)))  _Float16 v8h;
typedef __attribute__((ext_vector_type(4)))  _Float16 v4h;
typedef __attribute__((ext_vector_type(8)))  float    v8f;
typedef __attribute__((ext_vector_type(2)))  float    v2f;
typedef __attribute__((ext_vector_type(4)))  int      v4i;
typedef __attribute__((ext_vector_type(8)))  int      v8i;
typedef __attribute__((ext_vector_type(4)))  unsigned int v4u;
typedef __attribute__((ext_vector_type(4)))  float    v4f;

#define BATCH 8
#define NN    2048
#define FF    256
#define ALPHA 0.2f
#define NEG_BIG (-9.0e15f)

// LDS stage: 256 rows (o) x 32 halfs (m) + 2-dword pad per row -> 36 halfs/row
#define ROWH   36                    // halfs per padded LDS row (72 B)
#define STAGEH (256 * ROWH)          // halfs per buffer (18432 B)

// ---------------------------------------------------------------------------
// Pass 1: Wh = h @ W (exact fp32 via V_WMMA_F32_16X16X4_F32), e1 = Wh·a1,
// e2 = Wh·a2, and WhT16[b][o][m] = (f16)Wh[b][m][o] via LDS transpose.
// ---------------------------------------------------------------------------
__global__ __launch_bounds__(32) void gat_pass1(
    const float* __restrict__ h, const float* __restrict__ W,
    const float* __restrict__ a, _Float16* __restrict__ whT,
    float* __restrict__ e1g, float* __restrict__ e2g)
{
  __shared__ _Float16 xl[256 * 18];   // stride 18 halves -> conflict-free banks
  const int lane = threadIdx.x;
  const int lc   = lane & 15;
  const int hi   = lane >> 4;
  const int blk  = blockIdx.x;
  const int b    = blk >> 7;
  const int it   = blk & 127;
  const int ibase = it << 4;
  const int row   = ibase + lc;
  const float* hrow = h + ((size_t)b * NN + row) * FF;

  v8f c[16] = {};

  for (int k = 0; k < 64; ++k) {
    const int kk = 4 * k + (hi ? 2 : 0);
    const float2 av = *(const float2*)(hrow + kk);
    v2f A; A.x = av.x; A.y = av.y;
#pragma unroll
    for (int ot = 0; ot < 16; ++ot) {
      v2f B;
      B.x = W[(size_t)kk       * FF + ot * 16 + lc];
      B.y = W[(size_t)(kk + 1) * FF + ot * 16 + lc];
      c[ot] = __builtin_amdgcn_wmma_f32_16x16x4_f32(
          false, A, false, B, (short)0, c[ot], false, false);
    }
  }

  float a1v[16], a2v[16];
#pragma unroll
  for (int t = 0; t < 16; ++t) {
    a1v[t] = a[t * 16 + lc];
    a2v[t] = a[256 + t * 16 + lc];
  }
  float s1[8] = {}, s2[8] = {};
#pragma unroll
  for (int t = 0; t < 16; ++t)
#pragma unroll
    for (int v = 0; v < 8; ++v) {
      s1[v] += c[t][v] * a1v[t];
      s2[v] += c[t][v] * a2v[t];
    }
#pragma unroll
  for (int v = 0; v < 8; ++v) {
    float r1 = s1[v], r2 = s2[v];
#pragma unroll
    for (int d = 1; d < 16; d <<= 1) {
      r1 += __shfl_xor(r1, d);
      r2 += __shfl_xor(r2, d);
    }
    if (lc == 0) {
      e1g[b * NN + ibase + v + 8 * hi] = r1;
      e2g[b * NN + ibase + v + 8 * hi] = r2;
    }
  }

#pragma unroll
  for (int t = 0; t < 16; ++t)
#pragma unroll
    for (int v = 0; v < 8; ++v)
      xl[(t * 16 + lc) * 18 + v + 8 * hi] = (_Float16)c[t][v];
  __syncthreads();
#pragma unroll
  for (int q = 0; q < 8; ++q) {
    const int o = lane + 32 * q;
    const _Float16* src = xl + o * 18;
    v16h val;
#pragma unroll
    for (int e = 0; e < 16; ++e) val[e] = src[e];
    *(v16h*)(whT + (size_t)(b * 256 + o) * NN + ibase) = val;
  }
}

// ---------------------------------------------------------------------------
// TDM: async-load one 256(o) x 32(m) f16 tile of WhT into LDS with 2-dword
// row padding (LDS row stride 72 B -> bank-conflict-free ds reads).
// ---------------------------------------------------------------------------
__device__ __forceinline__ void tdm_load_tile(const _Float16* gsrc,
                                              unsigned lds_off) {
  const unsigned long long ga = (unsigned long long)(size_t)gsrc;
  v4u g0;
  g0.x = 1u;                                         // count=1, user descriptor
  g0.y = lds_off;                                    // lds_addr (bytes)
  g0.z = (unsigned)(ga & 0xFFFFFFFFu);               // global_addr[31:0]
  g0.w = (unsigned)((ga >> 32) & 0x01FFFFFFu)        // global_addr[56:32]
         | (2u << 30);                               // type=2 (image)
  const v8i g1 = {
      (int)((1u << 16)       // data_size = 2 B
            | (1u << 20)     // pad_enable
            | (3u << 22)     // pad_interval: 16 dwords (one 64 B row)
            | (1u << 25)),   // pad_amount: 2 dwords (8 B)
      (int)(((unsigned)NN & 0xFFFFu) << 16),   // tensor_dim0 = 2048 (m)
      (int)(256u << 16),                       // tensor_dim1 = 256 (o)
      (int)(32u << 16),                        // tile_dim0 = 32
      (int)256,                                // tile_dim1 = 256
      (int)NN,                                 // tensor_dim0_stride = 2048
      0, 0};
  const v4i z4 = {0, 0, 0, 0};
#if defined(__clang_major__) && (__clang_major__ >= 23)
  const v8i z8 = {0, 0, 0, 0, 0, 0, 0, 0};
  __builtin_amdgcn_tensor_load_to_lds(g0, g1, z4, z4, z8, 0);
#else
  __builtin_amdgcn_tensor_load_to_lds(g0, g1, z4, z4, 0);
#endif
}

// ---------------------------------------------------------------------------
// Pass 2: online-softmax GAT aggregation. 4 waves / workgroup share a
// double-buffered TDM-staged WhT tile; adj is read exactly once (HBM floor).
// P·Wh accumulated with V_WMMA_F32_16X16X32_F16 (fp32 accumulate).
// ---------------------------------------------------------------------------
__global__ __launch_bounds__(128) void gat_pass2(
    const int* __restrict__ adj, const _Float16* __restrict__ whT,
    const float* __restrict__ e1g, const float* __restrict__ e2g,
    float* __restrict__ out)
{
  __shared__ __align__(16) _Float16 bstage[2][STAGEH];  // 2 x 18 KB

  const int tid  = threadIdx.x;
  const int lane = tid & 31;
  const int wv   = tid >> 5;            // wave 0..3
  const int lc   = lane & 15;
  const int hi   = lane >> 4;
  const int koff = hi << 3;
  const int b    = blockIdx.x >> 5;     // 32 workgroups per batch
  const int it   = blockIdx.x & 31;
  const int ibase = (it << 6) + (wv << 4);
  const int row   = ibase + lc;

  const int*      adjrow = adj + ((size_t)b * NN + row) * NN;
  const float*    e2b    = e2g + b * NN;
  const _Float16* whTb   = whT + (size_t)b * 256 * NN;
  const float     e1r    = e1g[b * NN + row];

  v8f c[16] = {};
  float M = NEG_BIG, denom = 0.0f;

  // Prologue: wave 0 kicks off the TDM for tile j=0 into buffer 0 (offset 0).
  if (tid < 32) tdm_load_tile(whTb + 0, 0u);

  for (int j = 0; j < NN; j += 32) {
    const int buf = (j >> 5) & 1;
    __syncthreads();                     // prior reads of buf^1 are complete
    if (tid < 32 && (j + 32) < NN)
      tdm_load_tile(whTb + j + 32,
                    (buf ^ 1) ? (unsigned)(STAGEH * 2) : 0u);
    // bstage is the sole LDS object -> buffer0 @ byte 0, buffer1 @ 2*STAGEH

    const int m0 = j + koff;
    if (j + 32 < NN) __builtin_prefetch(adjrow + m0 + 32, 0, 1);

    const v4i A0 = *(const v4i*)(adjrow + m0);
    const v4i A1 = *(const v4i*)(adjrow + m0 + 4);
    const v4i A2 = *(const v4i*)(adjrow + m0 + 16);
    const v4i A3 = *(const v4i*)(adjrow + m0 + 20);
    const v4f F0 = *(const v4f*)(e2b + m0);
    const v4f F1 = *(const v4f*)(e2b + m0 + 4);
    const v4f F2 = *(const v4f*)(e2b + m0 + 16);
    const v4f F3 = *(const v4f*)(e2b + m0 + 20);

    float s[16];
#pragma unroll
    for (int e = 0; e < 4; ++e) {
      float x;
      x = e1r + F0[e]; x = x > 0.0f ? x : ALPHA * x; s[e]      = A0[e] > 0 ? x : NEG_BIG;
      x = e1r + F1[e]; x = x > 0.0f ? x : ALPHA * x; s[e + 4]  = A1[e] > 0 ? x : NEG_BIG;
      x = e1r + F2[e]; x = x > 0.0f ? x : ALPHA * x; s[e + 8]  = A2[e] > 0 ? x : NEG_BIG;
      x = e1r + F3[e]; x = x > 0.0f ? x : ALPHA * x; s[e + 12] = A3[e] > 0 ? x : NEG_BIG;
    }

    float lm = s[0];
#pragma unroll
    for (int e = 1; e < 16; ++e) lm = fmaxf(lm, s[e]);
    lm = fmaxf(lm, __shfl_xor(lm, 16));
    const float Mn    = fmaxf(M, lm);
    const float scale = __expf(M - Mn);
    M = Mn;

    float ls = 0.0f;
    v16h Av;
#pragma unroll
    for (int e = 0; e < 16; ++e) {
      const float p = __expf(s[e] - Mn);
      ls += p;
      Av[e] = (_Float16)p;               // element e <-> K=(e<8?e:e+8)+koff
    }
    ls += __shfl_xor(ls, 16);
    denom = denom * scale + ls;

    if (__any(scale < 1.0f)) {
#pragma unroll
      for (int v = 0; v < 8; ++v) {
        const float sv = __shfl(scale, v + koff);
#pragma unroll
        for (int t = 0; t < 16; ++t) c[t][v] *= sv;
      }
    }

    if (tid < 32) {
      if (j + 32 < NN) __builtin_amdgcn_s_wait_tensorcnt(1);
      else             __builtin_amdgcn_s_wait_tensorcnt(0);
    }
    __syncthreads();                     // staged tile j is visible to all

    const _Float16* bs = &bstage[buf][0];
#pragma unroll
    for (int ot = 0; ot < 16; ++ot) {
      // B layout: lane -> N=lc; K = 8*hi + 0..7 and 16 + 8*hi + 0..7 in padded row
      const _Float16* p = bs + (ot * 16 + lc) * ROWH + hi * 8;
      const v4h b0 = *(const v4h*)(p);
      const v4h b1 = *(const v4h*)(p + 4);
      const v4h b2 = *(const v4h*)(p + 16);
      const v4h b3 = *(const v4h*)(p + 20);
      const v8h lo = __builtin_shufflevector(b0, b1, 0, 1, 2, 3, 4, 5, 6, 7);
      const v8h hi8 = __builtin_shufflevector(b2, b3, 0, 1, 2, 3, 4, 5, 6, 7);
      const v16h Bv = __builtin_shufflevector(lo, hi8, 0, 1, 2, 3, 4, 5, 6, 7,
                                              8, 9, 10, 11, 12, 13, 14, 15);
      c[ot] = __builtin_amdgcn_wmma_f32_16x16x32_f16(
          false, Av, false, Bv, (short)0, c[ot], false, false);
    }
  }

  const float inv = 1.0f / denom;
  float dv[8];
#pragma unroll
  for (int v = 0; v < 8; ++v) dv[v] = __shfl(inv, v + koff);
#pragma unroll
  for (int t = 0; t < 16; ++t)
#pragma unroll
    for (int v = 0; v < 8; ++v) {
      float x = c[t][v] * dv[v];
      x = x > 0.0f ? x : (__expf(x) - 1.0f);
      out[((size_t)b * NN + ibase + v + 8 * hi) * FF + t * 16 + lc] = x;
    }
}

extern "C" void kernel_launch(void* const* d_in, const int* in_sizes, int n_in,
                              void* d_out, int out_size, void* d_ws, size_t ws_size,
                              hipStream_t stream) {
  const float* h   = (const float*)d_in[0];
  const int*   adj = (const int*)d_in[1];
  const float* W   = (const float*)d_in[2];
  const float* a   = (const float*)d_in[3];
  float* out = (float*)d_out;

  char* ws = (char*)d_ws;
  _Float16* whT = (_Float16*)ws;                                  // 8 MB
  float* e1g = (float*)(ws + (size_t)BATCH * 256 * NN * 2);       // 64 KB
  float* e2g = e1g + BATCH * NN;                                  // 64 KB

  hipLaunchKernelGGL(gat_pass1, dim3(BATCH * (NN / 16)), dim3(32), 0, stream,
                     h, W, a, whT, e1g, e2g);
  hipLaunchKernelGGL(gat_pass2, dim3(BATCH * (NN / 64)), dim3(128), 0, stream,
                     adj, whT, e1g, e2g, out);
}
